// RecKeyConv_64982855188921
// MI455X (gfx1250) — matrix operand order repaired and checked by reading, not compiled, not run on hardware
//
#include <hip/hip_runtime.h>
#include <math.h>
#include <stdint.h>

// ---------- problem constants ----------
#define BB 32
#define NR 4096
#define NK 128
#define NH 4
#define DD 128
#define IN_ 128
#define KNN 8
#define SCALE_INV_SQRT_D 0.08838834764831845f  // 1/sqrt(128)

// ---------- WMMA types ----------
typedef __attribute__((ext_vector_type(16))) __bf16 v16bf;
typedef __attribute__((ext_vector_type(8)))  float  v8f;

union Frag16 { v16bf v; uint4 q[2]; };

// round-to-nearest-even f32 -> bf16 (stored as u16)
static __device__ __forceinline__ unsigned short f2bf(float f) {
    unsigned int u = __float_as_uint(f);
    u = (u + 0x7FFFu + ((u >> 16) & 1u)) >> 16;
    return (unsigned short)u;
}

// ---------- 1) Gram matrices M[h] = W_h @ W_h^T, [4][128][128] ----------
__global__ void gram_kernel(const float* __restrict__ W, float* __restrict__ Mh) {
    int t = blockIdx.x * blockDim.x + threadIdx.x;      // 4*128*128
    if (t >= NH * IN_ * IN_) return;
    int i = t & 127, j = (t >> 7) & 127, h = t >> 14;
    const float* wj = W + (size_t)j * (NH * DD) + h * DD;
    const float* wi = W + (size_t)i * (NH * DD) + h * DD;
    float acc = 0.f;
    for (int d = 0; d < DD; ++d) acc = fmaf(wj[d], wi[d], acc);
    Mh[t] = acc;
}

// ---------- 2) G[b,h,k,i] = (1/sqrt(D)) * h0_kp[b,k,:] @ M[h][:,i], store bf16 ----------
__global__ void g_kernel(const float* __restrict__ h0, const float* __restrict__ Mh,
                         unsigned short* __restrict__ Gb) {
    int t = blockIdx.x * blockDim.x + threadIdx.x;      // 32*4*128*128
    if (t >= BB * NH * NK * IN_) return;
    int i = t & 127, k = (t >> 7) & 127, h = (t >> 14) & 3, b = t >> 16;
    const float* hp = h0 + ((size_t)b * NK + k) * IN_;
    const float* mp = Mh + (size_t)h * IN_ * IN_ + i;
    float acc = 0.f;
    for (int j = 0; j < IN_; ++j) acc = fmaf(hp[j], mp[j * IN_], acc);
    Gb[t] = f2bf(acc * SCALE_INV_SQRT_D);
}

// ---------- 3) h_rec f32 -> bf16 ----------
__global__ void cvt_kernel(const float* __restrict__ src, unsigned short* __restrict__ dst, int n) {
    int t = blockIdx.x * blockDim.x + threadIdx.x;
    if (t < n) dst[t] = f2bf(src[t]);
}

// ---------- 4) fused scores GEMM + online softmax + position aggregation ----------
// block = (b,h), 8 waves; wave w owns kp rows [w*16, w*16+16)
__global__ __launch_bounds__(256) void attn_kernel(
    const unsigned short* __restrict__ Gb,   // [B,H,NK,IN] bf16
    const unsigned short* __restrict__ Hb,   // [B,NR,IN] bf16
    const float* __restrict__ x_rec,         // [B,NR,3]
    float* __restrict__ part)                // [B,H,NK,3] per-head partial positions
{
    const int b    = blockIdx.x >> 2;
    const int h    = blockIdx.x & 3;
    const int wave = threadIdx.x >> 5;
    const int lane = threadIdx.x & 31;
    const int col  = lane & 15;
    const int hi   = lane >> 4;

    // A fragments: G rows (kp), 4 K-chunks of 32 held in registers for whole r loop.
    // 16-bit A 16x32 layout: lanes<16 hold K {0..7,16..23}; lanes>=16 hold K {8..15,24..31}.
    const unsigned short* Grow =
        Gb + (((size_t)(b * NH + h) * NK) + wave * 16 + col) * IN_;
    Frag16 A[4];
#pragma unroll
    for (int c = 0; c < 4; ++c) {
        const unsigned short* p = Grow + c * 32 + hi * 8;
        A[c].q[0] = *(const uint4*)(p);
        A[c].q[1] = *(const uint4*)(p + 16);
    }

    float m[8], s[8], px[8], py[8], pz[8];
#pragma unroll
    for (int g = 0; g < 8; ++g) { m[g] = -1e30f; s[g] = 0.f; px[g] = 0.f; py[g] = 0.f; pz[g] = 0.f; }

    const unsigned short* Hbase = Hb + (size_t)b * NR * IN_;
    const float* xbase = x_rec + (size_t)b * NR * 3;

    for (int r0 = 0; r0 < NR; r0 += 16) {
        const int r = r0 + col;
        v8f C = {};
        // B fragments: 16-bit B 32x16 layout: lanes<16 hold K {0..7,16..23}; lanes>=16 hold K {8..15,24..31}.
        const unsigned short* Hrow = Hbase + (size_t)r * IN_ + hi * 8;
#pragma unroll
        for (int c = 0; c < 4; ++c) {
            Frag16 Bm;
            const unsigned short* p = Hrow + c * 32;
            Bm.q[0] = *(const uint4*)(p);
            Bm.q[1] = *(const uint4*)(p + 16);
            C = __builtin_amdgcn_wmma_f32_16x16x32_bf16(
                    false, A[c].v, false, Bm.v, (short)0, C, false, false);
        }
        const float xr = xbase[r * 3 + 0];
        const float yr = xbase[r * 3 + 1];
        const float zr = xbase[r * 3 + 2];
        // online softmax update: lane holds 8 rows (M = hi*8+g), one column (r)
#pragma unroll
        for (int g = 0; g < 8; ++g) {
            float sc   = C[g];
            float mn   = fmaxf(m[g], sc);
            float corr = __expf(m[g] - mn);
            float e    = __expf(sc - mn);
            s[g]  = s[g]  * corr + e;
            px[g] = px[g] * corr + e * xr;
            py[g] = py[g] * corr + e * yr;
            pz[g] = pz[g] * corr + e * zr;
            m[g]  = mn;
        }
    }

    // merge across the 16 lanes (columns) sharing each row; xor masks stay in lane-halves
#pragma unroll
    for (int off = 1; off <= 8; off <<= 1) {
#pragma unroll
        for (int g = 0; g < 8; ++g) {
            float m2 = __shfl_xor(m[g],  off, 32);
            float s2 = __shfl_xor(s[g],  off, 32);
            float x2 = __shfl_xor(px[g], off, 32);
            float y2 = __shfl_xor(py[g], off, 32);
            float z2 = __shfl_xor(pz[g], off, 32);
            float mn = fmaxf(m[g], m2);
            float e1 = __expf(m[g] - mn);
            float e2 = __expf(m2 - mn);
            s[g]  = s[g]  * e1 + s2 * e2;
            px[g] = px[g] * e1 + x2 * e2;
            py[g] = py[g] * e1 + y2 * e2;
            pz[g] = pz[g] * e1 + z2 * e2;
            m[g]  = mn;
        }
    }

    if (col == 0) {
#pragma unroll
        for (int g = 0; g < 8; ++g) {
            int k = wave * 16 + hi * 8 + g;
            float inv = 1.0f / (s[g] * (float)NH);   // softmax norm and mean over heads
            float* o = part + (((size_t)(b * NH + h) * NK) + k) * 3;
            o[0] = px[g] * inv;
            o[1] = py[g] * inv;
            o[2] = pz[g] * inv;
        }
    }
}

// ---------- 5) combine per-head partials (fixed order => deterministic) ----------
__global__ void combine_kernel(const float* __restrict__ part, float* __restrict__ kp_pos) {
    int t = blockIdx.x * blockDim.x + threadIdx.x;    // B*NK*3
    if (t >= BB * NK * 3) return;
    int c = t % 3, k = (t / 3) & 127, b = t / (NK * 3);
    float acc = 0.f;
#pragma unroll
    for (int h = 0; h < NH; ++h)
        acc += part[(((size_t)(b * NH + h) * NK) + k) * 3 + c];
    kp_pos[t] = acc;
}

// ---------- 6) kNN: top-8 nearest x0_rec per keypoint (jax top_k tie semantics) ----------
__global__ __launch_bounds__(32) void knn_kernel(const float* __restrict__ kp_pos,
                                                 const float* __restrict__ x0,
                                                 int* __restrict__ idx_out) {
    const int bk = blockIdx.x;                // b*NK + k
    const int b  = bk >> 7;
    const int lane = threadIdx.x;
    const float kx = kp_pos[(size_t)bk * 3 + 0];
    const float ky = kp_pos[(size_t)bk * 3 + 1];
    const float kz = kp_pos[(size_t)bk * 3 + 2];
    const float* xb = x0 + (size_t)b * NR * 3;

    float bd[KNN]; int bi[KNN];
#pragma unroll
    for (int j = 0; j < KNN; ++j) { bd[j] = 3.4e38f; bi[j] = 1 << 30; }

    for (int r = lane; r < NR; r += 32) {
        float dx = kx - xb[r * 3 + 0];
        float dy = ky - xb[r * 3 + 1];
        float dz = kz - xb[r * 3 + 2];
        float d  = dx * dx + dy * dy + dz * dz;   // monotonic in true distance
        bool ins = (d < bd[KNN - 1]) || (d == bd[KNN - 1] && r < bi[KNN - 1]);
        if (ins) { bd[KNN - 1] = d; bi[KNN - 1] = r; }
#pragma unroll
        for (int j = KNN - 1; j >= 1; --j) {
            float a0 = bd[j - 1], a1 = bd[j];
            int   i0 = bi[j - 1], i1 = bi[j];
            bool sw = (a1 < a0) || (a1 == a0 && i1 < i0);
            bd[j - 1] = sw ? a1 : a0; bd[j] = sw ? a0 : a1;
            bi[j - 1] = sw ? i1 : i0; bi[j] = sw ? i0 : i1;
        }
    }

    __shared__ float sd[32 * KNN];
    __shared__ int   si[32 * KNN];
#pragma unroll
    for (int j = 0; j < KNN; ++j) { sd[lane * KNN + j] = bd[j]; si[lane * KNN + j] = bi[j]; }
    __syncthreads();
    if (lane == 0) {
        int* out = idx_out + (size_t)bk * KNN;
        for (int j = 0; j < KNN; ++j) {
            float best = 3.4e38f; int bidx = 1 << 30; int bpos = 0;
            for (int t = 0; t < 32 * KNN; ++t) {
                if (sd[t] < best || (sd[t] == best && si[t] < bidx)) {
                    best = sd[t]; bidx = si[t]; bpos = t;
                }
            }
            out[j] = bidx;
            sd[bpos] = 3.4e38f;
        }
    }
}

// ---------- 7) gather + distance features + MLP + SiLU + LayerNorm ----------
__global__ __launch_bounds__(128) void mlp_kernel(
    const float* __restrict__ h_rec, const float* __restrict__ x_rec,
    const float* __restrict__ kp_pos, const int* __restrict__ idx,
    const float* __restrict__ Wm, const float* __restrict__ bm,
    const float* __restrict__ gamma, const float* __restrict__ beta,
    float* __restrict__ out_feat)
{
    const int bk = blockIdx.x;        // b*NK + k
    const int b  = bk >> 7;
    const int t  = threadIdx.x;       // output dim / feature index

    __shared__ float feat[DD + KNN];
    __shared__ int   nb[KNN];
    __shared__ float red[DD];

    if (t < KNN) nb[t] = idx[(size_t)bk * KNN + t];
    __syncthreads();

    {   // mean of neighbor features
        float acc = 0.f;
#pragma unroll
        for (int j = 0; j < KNN; ++j)
            acc += h_rec[((size_t)b * NR + nb[j]) * IN_ + t];
        feat[t] = acc * (1.0f / KNN);
    }
    if (t < KNN) {   // distances to keypoint (reference adds 1e-30 per component)
        const float* kp = kp_pos + (size_t)bk * 3;
        const float* xn = x_rec + ((size_t)b * NR + nb[t]) * 3;
        float dx = xn[0] - kp[0] + 1e-30f;
        float dy = xn[1] - kp[1] + 1e-30f;
        float dz = xn[2] - kp[2] + 1e-30f;
        feat[DD + t] = sqrtf(dx * dx + dy * dy + dz * dz);
    }
    __syncthreads();

    float acc = bm[t];
    for (int i = 0; i < DD + KNN; ++i)
        acc = fmaf(feat[i], Wm[(size_t)i * DD + t], acc);
    float v = acc / (1.0f + __expf(-acc));   // SiLU

    // LayerNorm over the 128 dims
    red[t] = v; __syncthreads();
    for (int st = 64; st > 0; st >>= 1) { if (t < st) red[t] += red[t + st]; __syncthreads(); }
    float mu = red[0] * (1.0f / DD);
    __syncthreads();
    red[t] = (v - mu) * (v - mu); __syncthreads();
    for (int st = 64; st > 0; st >>= 1) { if (t < st) red[t] += red[t + st]; __syncthreads(); }
    float var = red[0] * (1.0f / DD);

    out_feat[(size_t)bk * DD + t] = (v - mu) * rsqrtf(var + 1e-5f) * gamma[t] + beta[t];
}

// ---------- host launcher ----------
extern "C" void kernel_launch(void* const* d_in, const int* in_sizes, int n_in,
                              void* d_out, int out_size, void* d_ws, size_t ws_size,
                              hipStream_t stream) {
    (void)in_sizes; (void)n_in; (void)out_size; (void)ws_size;
    const float* h_rec  = (const float*)d_in[0];
    const float* h0_kp  = (const float*)d_in[1];
    const float* x_rec  = (const float*)d_in[2];
    const float* x0_rec = (const float*)d_in[3];
    const float* W_src  = (const float*)d_in[4];
    const float* W_mlp  = (const float*)d_in[5];
    const float* b_mlp  = (const float*)d_in[6];
    const float* ln_g   = (const float*)d_in[7];
    const float* ln_b   = (const float*)d_in[8];

    float* out     = (float*)d_out;
    float* kp_pos  = out;                       // [B,NK,3]   = 12288 floats
    float* kp_feat = out + BB * NK * 3;         // [B,NK,128] = 524288 floats

    char* ws = (char*)d_ws;
    size_t off = 0;
    float*          Mh   = (float*)(ws + off);          off += (size_t)NH * IN_ * IN_ * 4;          // 256 KB
    unsigned short* Gb   = (unsigned short*)(ws + off); off += (size_t)BB * NH * NK * IN_ * 2;      // 4 MB
    unsigned short* Hb   = (unsigned short*)(ws + off); off += (size_t)BB * NR * IN_ * 2;           // 33.5 MB
    int*            nidx = (int*)(ws + off);            off += (size_t)BB * NK * KNN * 4;           // 128 KB
    float*          part = (float*)(ws + off);          off += (size_t)BB * NH * NK * 3 * 4;        // 192 KB

    // 1) Gram matrices
    gram_kernel<<<(NH * IN_ * IN_ + 255) / 256, 256, 0, stream>>>(W_src, Mh);
    // 2) G (bf16, pre-scaled by 1/sqrt(D))
    g_kernel<<<(BB * NH * NK * IN_ + 255) / 256, 256, 0, stream>>>(h0_kp, Mh, Gb);
    // 3) h_rec -> bf16
    {
        int n = BB * NR * IN_;
        cvt_kernel<<<(n + 255) / 256, 256, 0, stream>>>(h_rec, Hb, n);
    }
    // 4) fused WMMA scores + softmax + position aggregation
    attn_kernel<<<BB * NH, 256, 0, stream>>>(Gb, Hb, x_rec, part);
    // 5) combine heads -> kp_pos (deterministic fixed-order sum)
    combine_kernel<<<(BB * NK * 3 + 255) / 256, 256, 0, stream>>>(part, kp_pos);
    // 6) kNN against x0_rec
    knn_kernel<<<BB * NK, 32, 0, stream>>>(kp_pos, x0_rec, nidx);
    // 7) gather + MLP + SiLU + LayerNorm -> kp_feat
    mlp_kernel<<<BB * NK, 128, 0, stream>>>(h_rec, x_rec, kp_pos, nidx,
                                            W_mlp, b_mlp, ln_g, ln_b, kp_feat);
}
